// GNNRecommender_31499290149338
// MI455X (gfx1250) — matrix operand order repaired and checked by reading, not compiled, not run on hardware
//
#include <hip/hip_runtime.h>
#include <hip/hip_bf16.h>
#include <hip/hip_fp16.h>

typedef __attribute__((ext_vector_type(16))) _Float16 v16h;
typedef __attribute__((ext_vector_type(8)))  _Float16 v8h;
typedef __attribute__((ext_vector_type(4)))  _Float16 v4h;
typedef __attribute__((ext_vector_type(8)))  float    v8f;

#define HDIM   128
#define BN_EPS 1e-5f

union AFrag { v16h v; v8h h[2]; };
union BFrag { v16h v; uint4 u[2]; };
union HCvt  { _Float16 f; unsigned short s; };

// ---------------------------------------------------------------------------
// Pack a 128x128 f32 weight matrix into per-lane WMMA B-fragment layout:
//   Bp[((kb*8 + nt)*32 + lane)*8 + r] = {W[k+1][n], W[k][n]} as packed f16,
//   k = kb*32 + (lane>>4)*16 + 2r, n = nt*16 + (lane&15)
// Contiguous k order, same as the A loads (lanes 0-15: K=0..15, lanes 16-31:
// K=16..31, two K values per VGPR). 8192 threads total.
// ---------------------------------------------------------------------------
__global__ void pack_weights(const float* __restrict__ W, unsigned* __restrict__ Bp) {
    int t = blockIdx.x * blockDim.x + threadIdx.x;
    if (t >= 4 * 8 * 32 * 8) return;
    int r    = t & 7;
    int lane = (t >> 3) & 31;
    int nt   = (t >> 8) & 7;
    int kb   = (t >> 11) & 3;
    int n = nt * 16 + (lane & 15);
    int k = kb * 32 + (lane >> 4) * 16 + 2 * r;
    HCvt lo, hi;
    lo.f = (_Float16)W[(size_t)k * HDIM + n];
    hi.f = (_Float16)W[(size_t)(k + 1) * HDIM + n];
    Bp[t] = ((unsigned)hi.s << 16) | (unsigned)lo.s;
}

__global__ void f32_to_f16_kernel(const float4* __restrict__ x, v4h* __restrict__ o, size_t n4) {
    size_t i = (size_t)blockIdx.x * blockDim.x + threadIdx.x;
    if (i >= n4) return;
    float4 a = x[i];
    v4h h;
    h.x = (_Float16)a.x; h.y = (_Float16)a.y; h.z = (_Float16)a.z; h.w = (_Float16)a.w;
    o[i] = h;
}

__global__ void fill_ones_kernel(float* __restrict__ p, int n) {
    int i = blockIdx.x * blockDim.x + threadIdx.x;
    if (i < n) p[i] = 1.0f;
}

__global__ void zero_kernel(float4* __restrict__ p, size_t n4) {
    size_t i = (size_t)blockIdx.x * blockDim.x + threadIdx.x;
    if (i < n4) p[i] = make_float4(0.f, 0.f, 0.f, 0.f);
}

__global__ void deg_accum_kernel(const int* __restrict__ dst, float* __restrict__ deg, int e) {
    int i = blockIdx.x * blockDim.x + threadIdx.x;
    if (i < e) atomicAdd(&deg[dst[i]], 1.0f);
}

__global__ void make_dinv_kernel(float* __restrict__ deg, int n) {
    int i = blockIdx.x * blockDim.x + threadIdx.x;
    if (i < n) deg[i] = rsqrtf(fmaxf(deg[i], 1.0f));
}

// ---------------------------------------------------------------------------
// WMMA GEMM: C[N,128] = A[N,128](f16) @ W(packed f16), f32 accumulate.
// One wave per 16-row tile; wave iterates all 8 N-tiles, A fragments for the
// whole K=128 kept in registers (4 x v16h). All 4 B fragments for a tile are
// loaded in one burst, then 4 WMMAs issue back-to-back.
// MODE 0: scale rows by dinv (GCN pre-norm). MODE 1: +bias, ReLU (classifier).
// ---------------------------------------------------------------------------
template<int MODE>
__global__ void wmma_gemm128(const _Float16* __restrict__ A,
                             const unsigned* __restrict__ Bp,
                             const float* __restrict__ scaleRow,
                             const float* __restrict__ bias,
                             float* __restrict__ Cout,
                             int Mtiles) {
    int lane  = threadIdx.x & 31;
    int wid   = threadIdx.x >> 5;
    int mtile = blockIdx.x * (blockDim.x >> 5) + wid;
    if (mtile >= Mtiles) return;              // wave-uniform: EXEC stays all-1s

    int mr = lane & 15;                       // A row / C column within tile
    int hh = lane >> 4;                       // lane half selects K/M halves

    const _Float16* Arow = A + ((size_t)mtile * 16 + mr) * HDIM;
    AFrag a[4];
#pragma unroll
    for (int kb = 0; kb < 4; ++kb) {
        const v8h* p = (const v8h*)(Arow + kb * 32 + 16 * hh);
        a[kb].h[0] = p[0];                    // K = kb*32 + 16h + 0..7
        a[kb].h[1] = p[1];                    // K = kb*32 + 16h + 8..15
    }

    float rowScale[8];
    if (MODE == 0) {
#pragma unroll
        for (int r = 0; r < 8; ++r)
            rowScale[r] = scaleRow[(size_t)mtile * 16 + r + 8 * hh];
    }

    for (int nt = 0; nt < 8; ++nt) {
        // burst-load all 4 K-block B fragments (L1-resident packed weights)
        BFrag b[4];
#pragma unroll
        for (int kb = 0; kb < 4; ++kb) {
            const uint4* bp = (const uint4*)(Bp + ((size_t)((kb * 8 + nt) * 32 + lane)) * 8);
            b[kb].u[0] = bp[0];
            b[kb].u[1] = bp[1];
        }
        v8f c = {};
#pragma unroll
        for (int kb = 0; kb < 4; ++kb)
            c = __builtin_amdgcn_wmma_f32_16x16x32_f16(
                    false, a[kb].v, false, b[kb].v, (short)0, c, false, false);

        int n = nt * 16 + mr;
        float badd = (MODE == 1) ? bias[n] : 0.0f;
#pragma unroll
        for (int r = 0; r < 8; ++r) {
            size_t row = (size_t)mtile * 16 + r + 8 * hh;   // C layout: VGPR r -> M=r+8h
            float d = c[r];
            if (MODE == 0) d *= rowScale[r];
            else           d = fmaxf(d + badd, 0.0f);
            Cout[row * HDIM + n] = d;
        }
    }
}

// One wave per edge: gather 128 contiguous floats of the source row (float4
// per lane), atomically accumulate into the destination row. acc (51.2 MB)
// and hw (51.2 MB) both fit in the 192 MB L2, so this runs at L2 rate.
__global__ void edge_scatter_kernel(const float* __restrict__ hw,
                                    const int* __restrict__ src,
                                    const int* __restrict__ dst,
                                    float* __restrict__ acc, int nE) {
    unsigned gt  = blockIdx.x * blockDim.x + threadIdx.x;
    int wid  = (int)(gt >> 5);
    int lane = threadIdx.x & 31;
    if (wid >= nE) return;
    int s = src[wid];
    int d = dst[wid];
    const float4* ps = (const float4*)(hw + (size_t)s * HDIM);
    float4 v = ps[lane];
    float* pd = acc + (size_t)d * HDIM + lane * 4;
    atomicAdd(pd + 0, v.x);
    atomicAdd(pd + 1, v.y);
    atomicAdd(pd + 2, v.z);
    atomicAdd(pd + 3, v.w);
}

// Fused: dst-side dinv scale + self-loop + bias + eval BatchNorm + ReLU,
// emitting f16 for the next WMMA GEMM's A operand. float4-vectorized.
__global__ void bn_relu_finalize(const float4* __restrict__ acc4, const float4* __restrict__ hw4,
                                 const float* __restrict__ dinv,
                                 const float* __restrict__ b,  const float* __restrict__ g,
                                 const float* __restrict__ be, const float* __restrict__ m,
                                 const float* __restrict__ v,
                                 v4h* __restrict__ out, size_t total4) {
    size_t i = (size_t)blockIdx.x * blockDim.x + threadIdx.x;
    if (i >= total4) return;
    int    q    = (int)(i & 31);              // which float4 within the 128-wide row
    size_t node = i >> 5;
    float4 av = acc4[i];
    float4 hv = hw4[i];
    float  di = dinv[node];
    float4 bv  = ((const float4*)b)[q];
    float4 gv  = ((const float4*)g)[q];
    float4 bev = ((const float4*)be)[q];
    float4 mv  = ((const float4*)m)[q];
    float4 vv  = ((const float4*)v)[q];
    v4h o;
    {
        float val = (av.x + hv.x) * di + bv.x;
        o.x = (_Float16)fmaxf(gv.x * (val - mv.x) * rsqrtf(vv.x + BN_EPS) + bev.x, 0.0f);
    }
    {
        float val = (av.y + hv.y) * di + bv.y;
        o.y = (_Float16)fmaxf(gv.y * (val - mv.y) * rsqrtf(vv.y + BN_EPS) + bev.y, 0.0f);
    }
    {
        float val = (av.z + hv.z) * di + bv.z;
        o.z = (_Float16)fmaxf(gv.z * (val - mv.z) * rsqrtf(vv.z + BN_EPS) + bev.z, 0.0f);
    }
    {
        float val = (av.w + hv.w) * di + bv.w;
        o.w = (_Float16)fmaxf(gv.w * (val - mv.w) * rsqrtf(vv.w + BN_EPS) + bev.w, 0.0f);
    }
    out[i] = o;
}

// y[i] = dot(z[i,:], cW2) + cb2 — one wave per node, shuffle reduction.
__global__ void out_reduce_kernel(const float* __restrict__ z, const float* __restrict__ cW2,
                                  const float* __restrict__ cb2, float* __restrict__ y, int n) {
    unsigned gt  = blockIdx.x * blockDim.x + threadIdx.x;
    int wid  = (int)(gt >> 5);
    int lane = threadIdx.x & 31;
    if (wid >= n) return;
    const float4* pz = (const float4*)(z + (size_t)wid * HDIM);
    const float4* pw = (const float4*)cW2;
    float4 a = pz[lane];
    float4 w = pw[lane];
    float s = a.x * w.x + a.y * w.y + a.z * w.z + a.w * w.w;
    for (int off = 16; off > 0; off >>= 1) s += __shfl_down(s, off, 32);
    if (lane == 0) y[wid] = s + cb2[0];
}

// ---------------------------------------------------------------------------
extern "C" void kernel_launch(void* const* d_in, const int* in_sizes, int n_in,
                              void* d_out, int out_size, void* d_ws, size_t ws_size,
                              hipStream_t stream) {
    const float* x     = (const float*)d_in[0];
    const int*   edges = (const int*)d_in[1];
    const float* W1  = (const float*)d_in[2];
    const float* b1  = (const float*)d_in[3];
    const float* g1  = (const float*)d_in[4];
    const float* be1 = (const float*)d_in[5];
    const float* m1  = (const float*)d_in[6];
    const float* v1  = (const float*)d_in[7];
    const float* W2  = (const float*)d_in[8];
    const float* b2  = (const float*)d_in[9];
    const float* g2  = (const float*)d_in[10];
    const float* be2 = (const float*)d_in[11];
    const float* m2  = (const float*)d_in[12];
    const float* v2  = (const float*)d_in[13];
    const float* cW1 = (const float*)d_in[14];
    const float* cb1 = (const float*)d_in[15];
    const float* cW2 = (const float*)d_in[16];
    const float* cb2 = (const float*)d_in[17];

    const size_t N = (size_t)in_sizes[0] / HDIM;   // 100000
    const int    E = in_sizes[1] / 2;              // 1600000
    const int*   eSrc = edges;
    const int*   eDst = edges + E;
    float* y = (float*)d_out;

    // ---- workspace carve (256B-aligned) ----
    char*  wsb = (char*)d_ws;
    size_t off = 0;
    auto carve = [&](size_t bytes) -> void* {
        void* p = wsb + off;
        off = (off + bytes + 255) & ~(size_t)255;
        return p;
    };
    const size_t packBytes = 4 * 8 * 32 * 8 * sizeof(unsigned);   // 32 KB
    unsigned*  Bp1  = (unsigned*)carve(packBytes);
    unsigned*  Bp2  = (unsigned*)carve(packBytes);
    unsigned*  Bp3  = (unsigned*)carve(packBytes);
    float*     dinv = (float*)carve(N * sizeof(float));
    _Float16*  Abuf = (_Float16*)carve(N * HDIM * sizeof(_Float16));
    float*     hw   = (float*)carve(N * HDIM * sizeof(float));
    float*     acc  = (float*)carve(N * HDIM * sizeof(float));

    const size_t NF     = N * HDIM;
    const size_t NF4    = NF / 4;
    const int    Mtiles = (int)((N + 15) / 16);
    const dim3   blk(256);

    // 0) pack weights into WMMA B-fragment layout
    pack_weights<<<32, blk, 0, stream>>>(W1,  Bp1);
    pack_weights<<<32, blk, 0, stream>>>(W2,  Bp2);
    pack_weights<<<32, blk, 0, stream>>>(cW1, Bp3);

    // 1) x -> f16
    f32_to_f16_kernel<<<(unsigned)((NF4 + 255) / 256), blk, 0, stream>>>(
        (const float4*)x, (v4h*)Abuf, NF4);

    // 2) degrees (self-loop => init 1), then dinv = rsqrt(deg)
    fill_ones_kernel<<<(unsigned)((N + 255) / 256), blk, 0, stream>>>(dinv, (int)N);
    deg_accum_kernel<<<(E + 255) / 256, blk, 0, stream>>>(eDst, dinv, E);
    make_dinv_kernel<<<(unsigned)((N + 255) / 256), blk, 0, stream>>>(dinv, (int)N);

    const unsigned gemmBlocks = (Mtiles + 7) / 8;
    const unsigned scatBlocks = (unsigned)(((size_t)E * 32 + 255) / 256);
    const unsigned nf4Blocks  = (unsigned)((NF4 + 255) / 256);

    // ---- GCN layer 1 ----
    zero_kernel<<<nf4Blocks, blk, 0, stream>>>((float4*)acc, NF4);
    wmma_gemm128<0><<<gemmBlocks, blk, 0, stream>>>(Abuf, Bp1, dinv, nullptr, hw, Mtiles);
    edge_scatter_kernel<<<scatBlocks, blk, 0, stream>>>(hw, eSrc, eDst, acc, E);
    bn_relu_finalize<<<nf4Blocks, blk, 0, stream>>>(
        (const float4*)acc, (const float4*)hw, dinv, b1, g1, be1, m1, v1, (v4h*)Abuf, NF4);

    // ---- GCN layer 2 ----
    zero_kernel<<<nf4Blocks, blk, 0, stream>>>((float4*)acc, NF4);
    wmma_gemm128<0><<<gemmBlocks, blk, 0, stream>>>(Abuf, Bp2, dinv, nullptr, hw, Mtiles);
    edge_scatter_kernel<<<scatBlocks, blk, 0, stream>>>(hw, eSrc, eDst, acc, E);
    bn_relu_finalize<<<nf4Blocks, blk, 0, stream>>>(
        (const float4*)acc, (const float4*)hw, dinv, b2, g2, be2, m2, v2, (v4h*)Abuf, NF4);

    // ---- classifier: z = relu(h2 @ cW1 + cb1); y = z @ cW2 + cb2 ----
    wmma_gemm128<1><<<gemmBlocks, blk, 0, stream>>>(Abuf, Bp3, nullptr, cb1, hw, Mtiles);
    out_reduce_kernel<<<(unsigned)((N * 32 + 255) / 256), blk, 0, stream>>>(hw, cW2, cb2, y, (int)N);
}
